// MultiHeadAttention_67018669687091
// MI455X (gfx1250) — matrix operand order repaired and checked by reading, not compiled, not run on hardware
//
#include <hip/hip_runtime.h>
#include <hip/hip_bf16.h>

// ---------------------------------------------------------------------------
// MHA forward for MI455X (gfx1250, wave32, WMMA bf16 16x16x32, f32 accum).
//   1) convert x / Wqkv / Wout f32 -> bf16
//   2) GEMM1 (WMMA): qkv = x @ Wqkv^T + bqkv -> Q,K contiguous [B,S,E] bf16,
//      V transposed per head [B,H,D,S] bf16
//   3) flash attention per (b,h): double-buffered K/V tiles staged in LDS via
//      GLOBAL_LOAD_ASYNC_TO_LDS_B128 (ASYNCcnt, in-order drain), WMMA QK^T +
//      online softmax + WMMA P@V, writes attn_out [B,S,E] bf16
//   4) GEMM2 (WMMA): out = attn_out @ Wout^T + bout -> f32
// ---------------------------------------------------------------------------

typedef __attribute__((ext_vector_type(16))) __bf16 v16bf;
typedef __attribute__((ext_vector_type(8)))  __bf16 v8bf;
typedef __attribute__((ext_vector_type(8)))  float  v8f;

#define E_ 1024
#define H_ 16
#define D_ 64
#define S_ 2048
#define B_ 2
#define M_ (B_ * S_)   // 4096

__device__ __forceinline__ v8f wmma_bf16(v16bf a, v16bf b, v8f c) {
  return __builtin_amdgcn_wmma_f32_16x16x32_bf16(
      false, a, false, b, (short)0, c, false, false);
}

__device__ __forceinline__ v8f zero_v8f() {
  v8f z;
#pragma unroll
  for (int i = 0; i < 8; ++i) z[i] = 0.0f;
  return z;
}

// A-fragment (16x32 bf16): row = lane&15; elem i<8: K=(lane>>4)*8+i,
// elem i>=8: K=16+(lane>>4)*8+(i-8). Two 16B loads.
__device__ __forceinline__ v16bf load_frag_a(const __bf16* __restrict__ base,
                                             int ld, int row0, int k0, int lane) {
  const int m = lane & 15, kh = lane >> 4;
  const __bf16* p = base + (size_t)(row0 + m) * ld + k0 + kh * 8;
  v8bf lo = *reinterpret_cast<const v8bf*>(p);
  v8bf hi = *reinterpret_cast<const v8bf*>(p + 16);
  v16bf a;
#pragma unroll
  for (int i = 0; i < 8; ++i) { a[i] = lo[i]; a[i + 8] = hi[i]; }
  return a;
}

// B-fragment from B^T stored row-major [N,K]: lane holds col N=lane&15;
// elem i<8: K=(lane>>4)*8+i, elem i>=8: K=16+(lane>>4)*8+(i-8). Two 16B loads.
__device__ __forceinline__ v16bf load_frag_b(const __bf16* __restrict__ base,
                                             int ld, int n0, int k0, int lane) {
  const int n = lane & 15, kh = lane >> 4;
  const __bf16* p = base + (size_t)(n0 + n) * ld + k0 + kh * 8;
  v8bf lo = *reinterpret_cast<const v8bf*>(p);
  v8bf hi = *reinterpret_cast<const v8bf*>(p + 16);
  v16bf b;
#pragma unroll
  for (int i = 0; i < 8; ++i) { b[i] = lo[i]; b[i + 8] = hi[i]; }
  return b;
}

// ---------------------------------------------------------------------------
__global__ void f32_to_bf16_kernel(const float* __restrict__ in,
                                   __bf16* __restrict__ out, int n) {
  int i = blockIdx.x * blockDim.x + threadIdx.x;
  if (i < n) out[i] = (__bf16)in[i];
}

// ---------------------------------------------------------------------------
// C[M,N] = A[M,K] @ Bw[N,K]^T + bias[N].  Block 64x128, 4 waves 2x2,
// wave tile 32x64 (8 WMMA per 6 fragment loads).
// MODE 0: f32 out.  MODE 1: qkv scatter epilogue.
template <int MODE>
__global__ __launch_bounds__(128) void gemm_bf16_kernel(
    const __bf16* __restrict__ A, const __bf16* __restrict__ Bw,
    const float* __restrict__ bias,
    float* __restrict__ Cf,
    __bf16* __restrict__ Qb, __bf16* __restrict__ Kb, __bf16* __restrict__ Vt,
    int M, int N, int K) {
  const int lane = threadIdx.x & 31;
  const int wave = threadIdx.x >> 5;
  const int wm = wave >> 1, wn = wave & 1;
  const int m_base = blockIdx.y * 64 + wm * 32;
  const int n_base = blockIdx.x * 128 + wn * 64;

  v8f acc[2][4];
#pragma unroll
  for (int i = 0; i < 2; ++i)
#pragma unroll
    for (int j = 0; j < 4; ++j) acc[i][j] = zero_v8f();

  for (int k0 = 0; k0 < K; k0 += 32) {
    v16bf a0 = load_frag_a(A, K, m_base, k0, lane);
    v16bf a1 = load_frag_a(A, K, m_base + 16, k0, lane);
    v16bf b0 = load_frag_b(Bw, K, n_base, k0, lane);
    v16bf b1 = load_frag_b(Bw, K, n_base + 16, k0, lane);
    v16bf b2 = load_frag_b(Bw, K, n_base + 32, k0, lane);
    v16bf b3 = load_frag_b(Bw, K, n_base + 48, k0, lane);
    acc[0][0] = wmma_bf16(a0, b0, acc[0][0]);
    acc[0][1] = wmma_bf16(a0, b1, acc[0][1]);
    acc[0][2] = wmma_bf16(a0, b2, acc[0][2]);
    acc[0][3] = wmma_bf16(a0, b3, acc[0][3]);
    acc[1][0] = wmma_bf16(a1, b0, acc[1][0]);
    acc[1][1] = wmma_bf16(a1, b1, acc[1][1]);
    acc[1][2] = wmma_bf16(a1, b2, acc[1][2]);
    acc[1][3] = wmma_bf16(a1, b3, acc[1][3]);
  }

  const int rl = lane & 15, hh = lane >> 4;
#pragma unroll
  for (int mi = 0; mi < 2; ++mi)
#pragma unroll
    for (int ni = 0; ni < 4; ++ni)
#pragma unroll
      for (int r = 0; r < 8; ++r) {
        const int row = m_base + mi * 16 + hh * 8 + r;
        const int col = n_base + ni * 16 + rl;
        const float v = acc[mi][ni][r] + bias[col];
        if (MODE == 0) {
          Cf[(size_t)row * N + col] = v;
        } else {
          const int sec = col >> 10;            // 0=Q 1=K 2=V
          const int cc = col & (E_ - 1);
          const int b = row >> 11;
          const int s = row & (S_ - 1);
          if (sec == 0) {
            Qb[((size_t)b * S_ + s) * E_ + cc] = (__bf16)v;
          } else if (sec == 1) {
            Kb[((size_t)b * S_ + s) * E_ + cc] = (__bf16)v;
          } else {
            const int i = s * E_ + cc;          // flat reinterpretation
            const int h = i >> 17;
            const int ss = (i >> 6) & (S_ - 1);
            const int d = i & (D_ - 1);
            Vt[(((size_t)b * H_ + h) * D_ + d) * S_ + ss] = (__bf16)v;
          }
        }
      }
}

// ---------------------------------------------------------------------------
// Flash attention. grid=(B*H, S/64), 128 threads = 4 waves; each wave owns a
// 16-row query tile. Uniform key-block trip count across waves; the workgroup
// cooperatively stages K (32x64) and V^T (64x32) tiles into double-buffered
// LDS with async loads so block n+1's DMA overlaps block n's WMMA/softmax.
__global__ __launch_bounds__(128) void attn_kernel(
    const __bf16* __restrict__ Qb, const __bf16* __restrict__ Kb,
    const __bf16* __restrict__ Vt, __bf16* __restrict__ AO) {
  __shared__ __align__(16) __bf16 kt[2][32][64];     // K tiles  [key][d]
  __shared__ __align__(16) __bf16 vtile[2][64][32];  // V^T tiles [d][key]
  __shared__ __align__(16) __bf16 lds_p[4][16][32];  // per-wave P relayout

  const int tid = threadIdx.x;
  const int lane = tid & 31;
  const int wave = tid >> 5;
  const int bh = blockIdx.x;
  const int q0 = blockIdx.y * 64 + wave * 16;

  const __bf16* qptr = Qb + (size_t)bh * (S_ * D_);
  const __bf16* kptr = Kb + (size_t)bh * (S_ * D_);
  const __bf16* vptr = Vt + (size_t)bh * (D_ * S_);

  const int rl = lane & 15, hh = lane >> 4;

  // per-thread chunk coordinates for cooperative staging (4 async b128 / thread)
  const int kc0_row = (tid * 2) >> 3,       kc0_col = ((tid * 2) & 7) * 8;
  const int kc1_row = (tid * 2 + 1) >> 3,   kc1_col = ((tid * 2 + 1) & 7) * 8;
  const int vc0_row = (tid * 2) >> 2,       vc0_col = ((tid * 2) & 3) * 8;
  const int vc1_row = (tid * 2 + 1) >> 2,   vc1_col = ((tid * 2 + 1) & 3) * 8;

#define ISSUE_TILES(n0_, buf_)                                                  \
  do {                                                                          \
    unsigned l0 = (unsigned)(size_t)(&kt[(buf_)][kc0_row][kc0_col]);            \
    unsigned g0 = (unsigned)((((n0_) + kc0_row) * D_ + kc0_col) * 2);           \
    asm volatile("global_load_async_to_lds_b128 %0, %1, %2"                     \
                 :: "v"(l0), "v"(g0), "s"(kptr) : "memory");                    \
    unsigned l1 = (unsigned)(size_t)(&kt[(buf_)][kc1_row][kc1_col]);            \
    unsigned g1 = (unsigned)((((n0_) + kc1_row) * D_ + kc1_col) * 2);           \
    asm volatile("global_load_async_to_lds_b128 %0, %1, %2"                     \
                 :: "v"(l1), "v"(g1), "s"(kptr) : "memory");                    \
    unsigned l2 = (unsigned)(size_t)(&vtile[(buf_)][vc0_row][vc0_col]);         \
    unsigned g2 = (unsigned)((vc0_row * S_ + (n0_) + vc0_col) * 2);             \
    asm volatile("global_load_async_to_lds_b128 %0, %1, %2"                     \
                 :: "v"(l2), "v"(g2), "s"(vptr) : "memory");                    \
    unsigned l3 = (unsigned)(size_t)(&vtile[(buf_)][vc1_row][vc1_col]);         \
    unsigned g3 = (unsigned)((vc1_row * S_ + (n0_) + vc1_col) * 2);             \
    asm volatile("global_load_async_to_lds_b128 %0, %1, %2"                     \
                 :: "v"(l3), "v"(g3), "s"(vptr) : "memory");                    \
  } while (0)

  const v16bf qa0 = load_frag_a(qptr, D_, q0, 0, lane);
  const v16bf qa1 = load_frag_a(qptr, D_, q0, 32, lane);

  v8f o[4];
#pragma unroll
  for (int i = 0; i < 4; ++i) o[i] = zero_v8f();
  float mrow[8], lrow[8];
#pragma unroll
  for (int r = 0; r < 8; ++r) { mrow[r] = -__builtin_inff(); lrow[r] = 0.0f; }

  const int nblocks = blockIdx.y * 2 + 2;   // uniform across all 4 waves
  ISSUE_TILES(0, 0);                        // prologue: block 0 -> buffer 0

  for (int blk = 0; blk < nblocks; ++blk) {
    const int n0 = blk * 32;
    const int buf = blk & 1;

    if (blk + 1 < nblocks) {
      // refill the other buffer (freed by previous iteration's barrier),
      // then drain only the current block's 4 loads (in-order completion)
      ISSUE_TILES(n0 + 32, buf ^ 1);
      asm volatile("s_wait_asynccnt 4" ::: "memory");
    } else {
      asm volatile("s_wait_asynccnt 0" ::: "memory");
    }
    __syncthreads();                        // current tiles visible to all

    if (n0 <= q0 + 15) {                    // wave-uniform causal predicate
      // S = Q(16x64) @ K^T(64x32), fragments read from LDS
      v8f s0 = zero_v8f(), s1 = zero_v8f();
      {
        v16bf kf0 = load_frag_b(&kt[buf][0][0], D_, 0, 0, lane);
        v16bf kf1 = load_frag_b(&kt[buf][0][0], D_, 16, 0, lane);
        s0 = wmma_bf16(qa0, kf0, s0);
        s1 = wmma_bf16(qa0, kf1, s1);
        kf0 = load_frag_b(&kt[buf][0][0], D_, 0, 32, lane);
        kf1 = load_frag_b(&kt[buf][0][0], D_, 16, 32, lane);
        s0 = wmma_bf16(qa1, kf0, s0);
        s1 = wmma_bf16(qa1, kf1, s1);
      }

      // scale + causal mask + online softmax (16-lane-half reductions)
#pragma unroll
      for (int r = 0; r < 8; ++r) {
        const int row = q0 + hh * 8 + r;
        float v0 = s0[r] * 0.125f;
        float v1 = s1[r] * 0.125f;
        if (n0 + rl > row)      v0 = -__builtin_inff();
        if (n0 + 16 + rl > row) v1 = -__builtin_inff();
        float bm = fmaxf(v0, v1);
#pragma unroll
        for (int off = 8; off > 0; off >>= 1) bm = fmaxf(bm, __shfl_xor(bm, off, 32));
        const float mnew = fmaxf(mrow[r], bm);
        const float alpha = __expf(mrow[r] - mnew);
        const float p0 = __expf(v0 - mnew);
        const float p1 = __expf(v1 - mnew);
        float ps = p0 + p1;
#pragma unroll
        for (int off = 8; off > 0; off >>= 1) ps += __shfl_xor(ps, off, 32);
        lrow[r] = lrow[r] * alpha + ps;
        mrow[r] = mnew;
        o[0][r] *= alpha; o[1][r] *= alpha; o[2][r] *= alpha; o[3][r] *= alpha;
        lds_p[wave][hh * 8 + r][rl]      = (__bf16)p0;
        lds_p[wave][hh * 8 + r][16 + rl] = (__bf16)p1;
      }
      // per-wave LDS fence (DS ops are in-order within a wave)
      asm volatile("s_wait_dscnt 0" ::: "memory");

      v16bf pf;
      {
        const __bf16* pp = &lds_p[wave][lane & 15][0];
        v8bf plo = *reinterpret_cast<const v8bf*>(pp + hh * 8);
        v8bf phi = *reinterpret_cast<const v8bf*>(pp + 16 + hh * 8);
#pragma unroll
        for (int i = 0; i < 8; ++i) { pf[i] = plo[i]; pf[i + 8] = phi[i]; }
      }

      // O += P(16x32) @ V(32x64) from the LDS V^T tile
#pragma unroll
      for (int di = 0; di < 4; ++di) {
        v16bf vf = load_frag_b(&vtile[buf][0][0], 32, di * 16, 0, lane);
        o[di] = wmma_bf16(pf, vf, o[di]);
      }
    }
    __syncthreads();                 // free this buffer for the next refill
  }
#undef ISSUE_TILES

  const int b = bh >> 4, h = bh & 15;
#pragma unroll
  for (int di = 0; di < 4; ++di)
#pragma unroll
    for (int r = 0; r < 8; ++r) {
      const int s = q0 + hh * 8 + r;
      const int d = di * 16 + rl;
      const float val = o[di][r] / lrow[r];
      AO[((size_t)(b * S_ + s)) * E_ + h * D_ + d] = (__bf16)val;
    }
}

// ---------------------------------------------------------------------------
extern "C" void kernel_launch(void* const* d_in, const int* in_sizes, int n_in,
                              void* d_out, int out_size, void* d_ws, size_t ws_size,
                              hipStream_t stream) {
  (void)in_sizes; (void)n_in; (void)out_size; (void)ws_size;
  const float* x    = (const float*)d_in[0];
  const float* Wqkv = (const float*)d_in[1];
  const float* bqkv = (const float*)d_in[2];
  const float* Wout = (const float*)d_in[3];
  const float* bout = (const float*)d_in[4];
  float* out = (float*)d_out;

  char* ws = (char*)d_ws;
  const size_t n_x    = (size_t)M_ * E_;
  const size_t n_wqkv = (size_t)3 * E_ * E_;
  const size_t n_wout = (size_t)E_ * E_;
  __bf16* xb    = (__bf16*)ws;  ws += n_x * 2;
  __bf16* Wqkvb = (__bf16*)ws;  ws += n_wqkv * 2;
  __bf16* Woutb = (__bf16*)ws;  ws += n_wout * 2;
  __bf16* Qb    = (__bf16*)ws;  ws += n_x * 2;
  __bf16* Kb    = (__bf16*)ws;  ws += n_x * 2;
  __bf16* Vt    = (__bf16*)ws;  ws += n_x * 2;
  __bf16* AOb   = (__bf16*)ws;  ws += n_x * 2;   // ~48 MB total

  f32_to_bf16_kernel<<<(int)((n_x + 255) / 256), 256, 0, stream>>>(x, xb, (int)n_x);
  f32_to_bf16_kernel<<<(int)((n_wqkv + 255) / 256), 256, 0, stream>>>(Wqkv, Wqkvb, (int)n_wqkv);
  f32_to_bf16_kernel<<<(int)((n_wout + 255) / 256), 256, 0, stream>>>(Wout, Woutb, (int)n_wout);

  gemm_bf16_kernel<1><<<dim3(3 * E_ / 128, M_ / 64), 128, 0, stream>>>(
      xb, Wqkvb, bqkv, nullptr, Qb, Kb, Vt, M_, 3 * E_, E_);

  attn_kernel<<<dim3(B_ * H_, S_ / 64), 128, 0, stream>>>(Qb, Kb, Vt, AOb);

  gemm_bf16_kernel<0><<<dim3(E_ / 128, M_ / 64), 128, 0, stream>>>(
      AOb, Woutb, bout, out, nullptr, nullptr, nullptr, M_, E_, E_);
}
